// StockLSTM_GNN_39599598469911
// MI455X (gfx1250) — matrix operand order repaired and checked by reading, not compiled, not run on hardware
//
#include <hip/hip_runtime.h>
#include <math.h>

typedef __bf16 bf16;
typedef __attribute__((ext_vector_type(16))) __bf16 v16bf;
typedef __attribute__((ext_vector_type(8)))  __bf16 v8bf;
typedef __attribute__((ext_vector_type(8)))  float  v8f;

#define Nn 10000
#define Tt 60
#define Ff 16
#define HLc 256
#define G4 1024
#define GHc 8
#define GCc 64
#define GDc 512
#define FHc 512
#define Ec 320000
#define Etot (Ec + Nn)

// ---------------- WMMA fragment helpers ----------------
// A fragment (16x32 bf16, MxK): lanes hold row M=lane&15, K kb..kb+7 (v0-3)
// and K kb+8..kb+15 (v4-7) with kb = (lane>>4)*16. Row index is CLAMPED to
// M-1 (branch-free, always in-bounds; clamped rows feed only masked stores).
__device__ __forceinline__ v16bf load_a_frag(const bf16* __restrict__ A, int lda,
                                             int row0, int lane, int kt, int Mm1) {
  int r = row0 + (lane & 15);
  r = r < Mm1 ? r : Mm1;
  int kb = kt * 32 + ((lane >> 4) * 16);
  const bf16* p = A + (size_t)r * lda + kb;
  v8bf lo = *(const v8bf*)(p);
  v8bf hi = *(const v8bf*)(p + 8);
  v16bf a;
#pragma unroll
  for (int i = 0; i < 8; ++i) { a[i] = lo[i]; a[8 + i] = hi[i]; }
  return a;
}

// B fragments are pre-packed so each lane's 16 bf16 values are contiguous:
// tile (kt,nt): lane l holds B[kt*32 + (l>>4)*16 + s, nt*16 + (l&15)], s=0..15.
__device__ __forceinline__ v16bf load_b_frag(const bf16* __restrict__ Bp, int NT,
                                             int kt, int nt, int lane) {
  return *(const v16bf*)(Bp + ((((size_t)kt * NT + nt) * 32 + lane) << 4));
}

__device__ __forceinline__ v8f wmma_bf16(v16bf a, v16bf b, v8f c) {
  return __builtin_amdgcn_wmma_f32_16x16x32_bf16(false, a, false, b, (short)0, c,
                                                 false, false);
}

__device__ __forceinline__ float sigmoidf_(float x) { return 1.f / (1.f + __expf(-x)); }

struct Frags {
  v16bf a0, a1;
  v16bf b[4];
};

// ---------------- utility kernels ----------------
// x_seq [N,T,F] f32 -> [N,T,32] bf16, zero-padded K (so LSTM input tiles are full)
__global__ void cvt_x_pad(const float* __restrict__ x, bf16* __restrict__ y) {
  long total = (long)Nn * Tt * 32;
  for (long i = blockIdx.x * (long)blockDim.x + threadIdx.x; i < total;
       i += (long)gridDim.x * blockDim.x) {
    int f = (int)(i & 31);
    long nt = i >> 5;                       // n*T + t
    y[i] = (f < Ff) ? (bf16)x[nt * Ff + f] : (bf16)0.f;
  }
}

__global__ void fill_f32(float* __restrict__ p, float v, long n) {
  for (long i = blockIdx.x * (long)blockDim.x + threadIdx.x; i < n;
       i += (long)gridDim.x * blockDim.x)
    p[i] = v;
}

__global__ void fill_bf16(bf16* __restrict__ p, float v, long n) {
  for (long i = blockIdx.x * (long)blockDim.x + threadIdx.x; i < n;
       i += (long)gridDim.x * blockDim.x)
    p[i] = (bf16)v;
}

__global__ void add_vec(const float* __restrict__ a, const float* __restrict__ b,
                        float* __restrict__ o, int n) {
  int i = blockIdx.x * blockDim.x + threadIdx.x;
  if (i < n) o[i] = a[i] + b[i];
}

// Pack B[k,n] = W[n,k]  (W is [Nout, Kin] f32) into WMMA tile layout, K padded.
__global__ void pack_wT(const float* __restrict__ W, bf16* __restrict__ Bp,
                        int Kin, int Nout, int KT) {
  int NT = Nout >> 4;
  long total = (long)KT * NT * 512;
  for (long i = blockIdx.x * (long)blockDim.x + threadIdx.x; i < total;
       i += (long)gridDim.x * blockDim.x) {
    int s = (int)(i & 15);
    int l = (int)((i >> 4) & 31);
    long tile = i >> 9;
    int nt = (int)(tile % NT), kt = (int)(tile / NT);
    int n = nt * 16 + (l & 15);
    int k = kt * 32 + ((l >> 4) * 16) + s;
    float v = (k < Kin) ? W[(size_t)n * Kin + k] : 0.f;
    Bp[i] = (bf16)v;
  }
}

// Pack B[k,n] = W[k,n]  (W is [Kin, Nout] f32).
__global__ void pack_wN(const float* __restrict__ W, bf16* __restrict__ Bp,
                        int Kin, int Nout, int KT) {
  int NT = Nout >> 4;
  long total = (long)KT * NT * 512;
  for (long i = blockIdx.x * (long)blockDim.x + threadIdx.x; i < total;
       i += (long)gridDim.x * blockDim.x) {
    int s = (int)(i & 15);
    int l = (int)((i >> 4) & 31);
    long tile = i >> 9;
    int nt = (int)(tile % NT), kt = (int)(tile / NT);
    int n = nt * 16 + (l & 15);
    int k = kt * 32 + ((l >> 4) * 16) + s;
    float v = (k < Kin) ? W[(size_t)k * Nout + n] : 0.f;
    Bp[i] = (bf16)v;
  }
}

__global__ void sum_f32(const float* __restrict__ x, long n, float* __restrict__ out) {
  float s = 0.f;
  for (long i = blockIdx.x * (long)blockDim.x + threadIdx.x; i < n;
       i += (long)gridDim.x * blockDim.x)
    s += x[i];
#pragma unroll
  for (int m = 16; m > 0; m >>= 1) s += __shfl_xor(s, m, 32);
  if ((threadIdx.x & 31) == 0)
    __hip_atomic_fetch_add(out, s, __ATOMIC_RELAXED, __HIP_MEMORY_SCOPE_AGENT);
}

__global__ void scale_one(float* p, float f) {
  if (threadIdx.x == 0 && blockIdx.x == 0) p[0] *= f;
}

// ---------------- generic WMMA GEMM: C = act(A*Bp + bias) ----------------
// per-wave 32x64 tile, workgroup = 8 waves (2 M-sub x 4 N-sub) -> 64x256 per WG
// KT is a template parameter -> full unroll + 2-stage software pipeline.
template <int KT>
__global__ void __launch_bounds__(256) gemm_bias_act(
    const bf16* __restrict__ A, int lda, int M,
    const bf16* __restrict__ Bp, int NT,
    const float* __restrict__ bias,
    float* __restrict__ Cf, bf16* __restrict__ Cb, int Nc, int act) {
  int wave = threadIdx.x >> 5, lane = threadIdx.x & 31;
  int row0 = blockIdx.x * 64 + (wave & 1) * 32;
  int col0 = blockIdx.y * 256 + (wave >> 1) * 64;
  int nt0  = col0 >> 4;
  int Mm1  = M - 1;
  v8f zero = {};
  v8f acc[2][4];
#pragma unroll
  for (int i = 0; i < 2; ++i)
#pragma unroll
    for (int j = 0; j < 4; ++j) acc[i][j] = zero;

  Frags f[2];
  // prologue: stage 0
  f[0].a0 = load_a_frag(A, lda, row0,      lane, 0, Mm1);
  f[0].a1 = load_a_frag(A, lda, row0 + 16, lane, 0, Mm1);
#pragma unroll
  for (int ns = 0; ns < 4; ++ns) f[0].b[ns] = load_b_frag(Bp, NT, 0, nt0 + ns, lane);

#pragma unroll
  for (int kt = 0; kt < KT; ++kt) {
    if (kt + 1 < KT) {                        // issue next tile's loads first
      Frags& n = f[(kt + 1) & 1];
      n.a0 = load_a_frag(A, lda, row0,      lane, kt + 1, Mm1);
      n.a1 = load_a_frag(A, lda, row0 + 16, lane, kt + 1, Mm1);
#pragma unroll
      for (int ns = 0; ns < 4; ++ns)
        n.b[ns] = load_b_frag(Bp, NT, kt + 1, nt0 + ns, lane);
    }
    Frags& c = f[kt & 1];
#pragma unroll
    for (int ns = 0; ns < 4; ++ns) {
      acc[0][ns] = wmma_bf16(c.a0, c.b[ns], acc[0][ns]);
      acc[1][ns] = wmma_bf16(c.a1, c.b[ns], acc[1][ns]);
    }
  }
  // C layout: lane holds col = (lane&15), rows v + 8*(lane>>4) per 16x16 tile
  int rbase = row0 + 8 * (lane >> 4);
  int cbase = col0 + (lane & 15);
#pragma unroll
  for (int ms = 0; ms < 2; ++ms)
#pragma unroll
    for (int ns = 0; ns < 4; ++ns) {
      int col = cbase + ns * 16;
      float bv = bias ? bias[col] : 0.f;
#pragma unroll
      for (int v = 0; v < 8; ++v) {
        int r = rbase + ms * 16 + v;
        if (r < M) {
          float val = acc[ms][ns][v] + bv;
          if (act == 1) val = val > 0.f ? val : 0.f;
          size_t idx = (size_t)r * Nc + col;
          if (Cf) Cf[idx] = val;
          if (Cb) Cb[idx] = (bf16)val;
        }
      }
    }
}

// ---------------- fused LSTM step ----------------
// g = h_prev * Whh^T + x * Wih^T + bias ; gate math ; update c ; h_out (bf16)
// Wave computes 32 rows x 16 j-cols across ALL FOUR gates (4 B frags / ktile).
// Unified K-loop over 8 recurrent tiles + KTx input tiles (compile-time select).
template <int KTx>
__device__ __forceinline__ void lstm_load_tile(
    int kt, const bf16* __restrict__ Hprev, const bf16* __restrict__ X, int ldx,
    const bf16* __restrict__ Bhh, const bf16* __restrict__ Bih,
    int row0, int jt, int lane, Frags& f) {
  const bf16* A; int lda; const bf16* B; int k;
  if (kt < 8) { A = Hprev; lda = HLc; B = Bhh; k = kt; }
  else        { A = X;     lda = ldx; B = Bih; k = kt - 8; }
  f.a0 = load_a_frag(A, lda, row0,      lane, k, Nn - 1);
  f.a1 = load_a_frag(A, lda, row0 + 16, lane, k, Nn - 1);
#pragma unroll
  for (int g = 0; g < 4; ++g) f.b[g] = load_b_frag(B, 64, k, g * 16 + jt, lane);
}

template <int KTx>
__global__ void __launch_bounds__(256) lstm_step(
    const bf16* __restrict__ Hprev,
    const bf16* __restrict__ X, int ldx,
    const bf16* __restrict__ Bhh, const bf16* __restrict__ Bih,
    const float* __restrict__ bias,
    float* __restrict__ Cst, bf16* __restrict__ Hout) {
  constexpr int KTT = 8 + KTx;
  int wave = threadIdx.x >> 5, lane = threadIdx.x & 31;
  int row0 = blockIdx.x * 64 + (wave & 1) * 32;
  int j0   = blockIdx.y * 64 + (wave >> 1) * 16;
  int jt   = j0 >> 4;
  v8f zero = {};
  v8f acc[2][4];
#pragma unroll
  for (int i = 0; i < 2; ++i)
#pragma unroll
    for (int j = 0; j < 4; ++j) acc[i][j] = zero;

  Frags f[2];
  lstm_load_tile<KTx>(0, Hprev, X, ldx, Bhh, Bih, row0, jt, lane, f[0]);
#pragma unroll
  for (int kt = 0; kt < KTT; ++kt) {
    if (kt + 1 < KTT)
      lstm_load_tile<KTx>(kt + 1, Hprev, X, ldx, Bhh, Bih, row0, jt, lane,
                          f[(kt + 1) & 1]);
    Frags& c = f[kt & 1];
#pragma unroll
    for (int g = 0; g < 4; ++g) {
      acc[0][g] = wmma_bf16(c.a0, c.b[g], acc[0][g]);
      acc[1][g] = wmma_bf16(c.a1, c.b[g], acc[1][g]);
    }
  }
  // epilogue: i,f,g,o aligned elementwise across acc[ms][0..3]
  int j = j0 + (lane & 15);
  float bi = bias[j], bff = bias[HLc + j], bg = bias[2 * HLc + j], bo = bias[3 * HLc + j];
  int rbase = row0 + 8 * (lane >> 4);
#pragma unroll
  for (int ms = 0; ms < 2; ++ms)
#pragma unroll
    for (int v = 0; v < 8; ++v) {
      int r = rbase + ms * 16 + v;
      if (r < Nn) {
        float it = sigmoidf_(acc[ms][0][v] + bi);
        float ft = sigmoidf_(acc[ms][1][v] + bff);
        float gt = tanhf(acc[ms][2][v] + bg);
        float ot = sigmoidf_(acc[ms][3][v] + bo);
        size_t idx = (size_t)r * HLc + j;
        float cn = ft * Cst[idx] + it * gt;
        Cst[idx] = cn;
        Hout[idx] = (bf16)(ot * tanhf(cn));
      }
    }
}

// ---------------- GAT edge kernels ----------------
__device__ __forceinline__ void edge_sd(int e, const int* __restrict__ ei,
                                        int& s, int& d) {
  if (e < Ec) { s = ei[e]; d = ei[Ec + e]; }
  else        { s = d = e - Ec; }
}

__global__ void gat_edge_logits(const int* __restrict__ ei, const float* __restrict__ ea,
                                const float* __restrict__ meanp,
                                const float* __restrict__ xl, const float* __restrict__ xr,
                                const float* __restrict__ we, const float* __restrict__ att,
                                float* __restrict__ logits, float* __restrict__ mx) {
  long t = blockIdx.x * (long)blockDim.x + threadIdx.x;
  if (t >= (long)Etot * GHc) return;
  int e = (int)(t >> 3), h = (int)(t & 7);
  int s, d; edge_sd(e, ei, s, d);
  float w = (e < Ec) ? ea[e] : meanp[0];
  const float* pl = xl + (size_t)s * GDc + h * GCc;
  const float* pr = xr + (size_t)d * GDc + h * GCc;
  const float* pw = we + h * GCc;
  const float* pa = att + h * GCc;
  float acc = 0.f;
#pragma unroll 4
  for (int c = 0; c < GCc; ++c) {
    float m = pl[c] + pr[c] + w * pw[c];
    m = m > 0.f ? m : 0.2f * m;                 // leaky_relu 0.2
    acc += m * pa[c];
  }
  logits[t] = acc;
  float* ad = mx + (size_t)d * GHc + h;         // float atomic-max trick
  if (acc >= 0.f) atomicMax((int*)ad, __float_as_int(acc));
  else            atomicMin((unsigned int*)ad, __float_as_uint(acc));
}

__global__ void gat_edge_exp(const int* __restrict__ ei, const float* __restrict__ logits,
                             const float* __restrict__ mx,
                             float* __restrict__ exb, float* __restrict__ den) {
  long t = blockIdx.x * (long)blockDim.x + threadIdx.x;
  if (t >= (long)Etot * GHc) return;
  int e = (int)(t >> 3), h = (int)(t & 7);
  int s, d; edge_sd(e, ei, s, d); (void)s;
  float ex = __expf(logits[t] - mx[(size_t)d * GHc + h]);
  exb[t] = ex;
  __hip_atomic_fetch_add(&den[(size_t)d * GHc + h], ex, __ATOMIC_RELAXED,
                         __HIP_MEMORY_SCOPE_AGENT);
}

__global__ void gat_scatter(const int* __restrict__ ei, const float* __restrict__ xl,
                            const float* __restrict__ exb, const float* __restrict__ den,
                            float* __restrict__ gacc) {
  long t = blockIdx.x * (long)blockDim.x + threadIdx.x;
  if (t >= (long)Etot * GHc) return;
  int e = (int)(t >> 3), h = (int)(t & 7);
  int s, d; edge_sd(e, ei, s, d);
  float a = exb[t] / den[(size_t)d * GHc + h];
  const float* pl = xl + (size_t)s * GDc + h * GCc;
  float* po = gacc + (size_t)d * GDc + h * GCc;
#pragma unroll 4
  for (int c = 0; c < GCc; ++c)
    __hip_atomic_fetch_add(&po[c], pl[c] * a, __ATOMIC_RELAXED,
                           __HIP_MEMORY_SCOPE_AGENT);
}

__global__ void gat_finish(const float* __restrict__ gacc, const float* __restrict__ gbias,
                           bf16* __restrict__ gout) {
  long i = blockIdx.x * (long)blockDim.x + threadIdx.x;
  if (i >= (long)Nn * GDc) return;
  int j = (int)(i % GDc);
  float v = gacc[i] + gbias[j];
  v = v > 0.f ? v : expm1f(v);                  // elu
  gout[i] = (bf16)v;
}

__global__ void concat_fused(const bf16* __restrict__ h, const bf16* __restrict__ g,
                             bf16* __restrict__ fused) {
  long total = (long)Nn * (HLc + GDc);
  for (long i = blockIdx.x * (long)blockDim.x + threadIdx.x; i < total;
       i += (long)gridDim.x * blockDim.x) {
    int n = (int)(i / (HLc + GDc)), j = (int)(i % (HLc + GDc));
    fused[i] = (j < HLc) ? h[(size_t)n * HLc + j] : g[(size_t)n * GDc + (j - HLc)];
  }
}

__global__ void fc2_reduce(const float* __restrict__ H, const float* __restrict__ w2,
                           const float* __restrict__ b2, float* __restrict__ out) {
  int n = blockIdx.x * (blockDim.x >> 5) + (threadIdx.x >> 5);
  int lane = threadIdx.x & 31;
  if (n >= Nn) return;
  float s = 0.f;
  for (int j = lane; j < FHc; j += 32) s += H[(size_t)n * FHc + j] * w2[j];
#pragma unroll
  for (int m = 16; m > 0; m >>= 1) s += __shfl_xor(s, m, 32);
  if (lane == 0) out[n] = s + b2[0];
}

// ---------------- host ----------------
extern "C" void kernel_launch(void* const* d_in, const int* in_sizes, int n_in,
                              void* d_out, int out_size, void* d_ws, size_t ws_size,
                              hipStream_t stream) {
  (void)in_sizes; (void)n_in; (void)out_size; (void)ws_size;
  const float* x_seq   = (const float*)d_in[0];
  const int*   ei      = (const int*)d_in[1];
  const float* ea      = (const float*)d_in[2];
  const float* w_ih0   = (const float*)d_in[3];
  const float* w_hh0   = (const float*)d_in[4];
  const float* b_ih0   = (const float*)d_in[5];
  const float* b_hh0   = (const float*)d_in[6];
  const float* w_ih1   = (const float*)d_in[7];
  const float* w_hh1   = (const float*)d_in[8];
  const float* b_ih1   = (const float*)d_in[9];
  const float* b_hh1   = (const float*)d_in[10];
  const float* g0_wl   = (const float*)d_in[11];
  const float* g0_bl   = (const float*)d_in[12];
  const float* g0_wr   = (const float*)d_in[13];
  const float* g0_br   = (const float*)d_in[14];
  const float* g0_we   = (const float*)d_in[15];
  const float* g0_att  = (const float*)d_in[16];
  const float* g0_bias = (const float*)d_in[17];
  const float* g1_wl   = (const float*)d_in[18];
  const float* g1_bl   = (const float*)d_in[19];
  const float* g1_wr   = (const float*)d_in[20];
  const float* g1_br   = (const float*)d_in[21];
  const float* g1_we   = (const float*)d_in[22];
  const float* g1_att  = (const float*)d_in[23];
  const float* g1_bias = (const float*)d_in[24];
  const float* fc1_w   = (const float*)d_in[25];
  const float* fc1_b   = (const float*)d_in[26];
  const float* fc2_w   = (const float*)d_in[27];
  const float* fc2_b   = (const float*)d_in[28];
  float* out = (float*)d_out;

  char* wp = (char*)d_ws;
  auto alloc = [&](size_t b) -> char* {
    char* p = wp; wp += (b + 255) & ~(size_t)255; return p;
  };
  bf16* x32    = (bf16*)alloc((size_t)Nn * Tt * 32 * 2);   // K-padded input
  bf16* Bhh0   = (bf16*)alloc((size_t)8 * 64 * 512 * 2);
  bf16* Bih0   = (bf16*)alloc((size_t)1 * 64 * 512 * 2);
  bf16* Bhh1   = (bf16*)alloc((size_t)8 * 64 * 512 * 2);
  bf16* Bih1   = (bf16*)alloc((size_t)8 * 64 * 512 * 2);
  float* bias0 = (float*)alloc(G4 * 4);
  float* bias1 = (float*)alloc(G4 * 4);
  bf16* h0a = (bf16*)alloc((size_t)Nn * HLc * 2);
  bf16* h0b = (bf16*)alloc((size_t)Nn * HLc * 2);
  bf16* h1a = (bf16*)alloc((size_t)Nn * HLc * 2);
  bf16* h1b = (bf16*)alloc((size_t)Nn * HLc * 2);
  float* c0 = (float*)alloc((size_t)Nn * HLc * 4);
  float* c1 = (float*)alloc((size_t)Nn * HLc * 4);
  bf16* g0_wlP = (bf16*)alloc((size_t)8 * 32 * 512 * 2);
  bf16* g0_wrP = (bf16*)alloc((size_t)8 * 32 * 512 * 2);
  bf16* g1_wlP = (bf16*)alloc((size_t)16 * 32 * 512 * 2);
  bf16* g1_wrP = (bf16*)alloc((size_t)16 * 32 * 512 * 2);
  bf16* fc1P   = (bf16*)alloc((size_t)24 * 32 * 512 * 2);
  float* xl    = (float*)alloc((size_t)Nn * GDc * 4);
  float* xr    = (float*)alloc((size_t)Nn * GDc * 4);
  float* logits= (float*)alloc((size_t)Etot * GHc * 4);
  float* exb   = (float*)alloc((size_t)Etot * GHc * 4);
  float* mx    = (float*)alloc((size_t)Nn * GHc * 4);
  float* den   = (float*)alloc((size_t)Nn * GHc * 4);
  float* gacc  = (float*)alloc((size_t)Nn * GDc * 4);
  bf16* g1b    = (bf16*)alloc((size_t)Nn * GDc * 2);
  bf16* g2b    = (bf16*)alloc((size_t)Nn * GDc * 2);
  bf16* fusedb = (bf16*)alloc((size_t)Nn * (HLc + GDc) * 2);
  float* relu1 = (float*)alloc((size_t)Nn * FHc * 4);
  float* meanbuf = (float*)alloc(256);

  // ---- prep: converts, weight packing, bias combine, state zeroing ----
  cvt_x_pad<<<4096, 256, 0, stream>>>(x_seq, x32);
  pack_wT<<<1024, 256, 0, stream>>>(w_hh0, Bhh0, HLc, G4, 8);
  pack_wT<<<256,  256, 0, stream>>>(w_ih0, Bih0, Ff,  G4, 1);
  pack_wT<<<1024, 256, 0, stream>>>(w_hh1, Bhh1, HLc, G4, 8);
  pack_wT<<<1024, 256, 0, stream>>>(w_ih1, Bih1, HLc, G4, 8);
  pack_wN<<<512,  256, 0, stream>>>(g0_wl, g0_wlP, HLc, GDc, 8);
  pack_wN<<<512,  256, 0, stream>>>(g0_wr, g0_wrP, HLc, GDc, 8);
  pack_wN<<<1024, 256, 0, stream>>>(g1_wl, g1_wlP, GDc, GDc, 16);
  pack_wN<<<1024, 256, 0, stream>>>(g1_wr, g1_wrP, GDc, GDc, 16);
  pack_wN<<<1024, 256, 0, stream>>>(fc1_w, fc1P, HLc + GDc, FHc, 24);
  add_vec<<<4, 256, 0, stream>>>(b_ih0, b_hh0, bias0, G4);
  add_vec<<<4, 256, 0, stream>>>(b_ih1, b_hh1, bias1, G4);
  fill_bf16<<<1024, 256, 0, stream>>>(h0a, 0.f, (long)Nn * HLc);
  fill_bf16<<<1024, 256, 0, stream>>>(h1a, 0.f, (long)Nn * HLc);
  fill_f32<<<2048, 256, 0, stream>>>(c0, 0.f, (long)Nn * HLc);
  fill_f32<<<2048, 256, 0, stream>>>(c1, 0.f, (long)Nn * HLc);
  fill_f32<<<1, 64, 0, stream>>>(meanbuf, 0.f, 1);
  sum_f32<<<512, 256, 0, stream>>>(ea, (long)Ec, meanbuf);
  scale_one<<<1, 64, 0, stream>>>(meanbuf, 1.f / (float)Ec);

  // ---- 2-layer LSTM, interleaved per timestep (no h1-sequence storage) ----
  const dim3 lgrid(157, 4);
  bf16 *h0p = h0a, *h0n = h0b, *h1p = h1a, *h1n = h1b;
  for (int t = 0; t < Tt; ++t) {
    lstm_step<1><<<lgrid, 256, 0, stream>>>(h0p, x32 + (size_t)t * 32, Tt * 32,
                                            Bhh0, Bih0, bias0, c0, h0n);
    lstm_step<8><<<lgrid, 256, 0, stream>>>(h1p, h0n, HLc,
                                            Bhh1, Bih1, bias1, c1, h1n);
    bf16* tp;
    tp = h0p; h0p = h0n; h0n = tp;
    tp = h1p; h1p = h1n; h1n = tp;
  }
  const bf16* node = h1p;  // last-written layer-1 hidden state [N, HL] bf16

  // ---- GATv2 layers ----
  long eth = (long)Etot * GHc;
  int eg = (int)((eth + 255) / 256);
  // layer 0: K = 256 (KT=8)
  gemm_bias_act<8><<<dim3(157, 2), 256, 0, stream>>>(node, HLc, Nn, g0_wlP, 32,
                                                     g0_bl, xl, (bf16*)nullptr, GDc, 0);
  gemm_bias_act<8><<<dim3(157, 2), 256, 0, stream>>>(node, HLc, Nn, g0_wrP, 32,
                                                     g0_br, xr, (bf16*)nullptr, GDc, 0);
  fill_f32<<<256, 256, 0, stream>>>(mx, -__builtin_inff(), (long)Nn * GHc);
  fill_f32<<<256, 256, 0, stream>>>(den, 0.f, (long)Nn * GHc);
  fill_f32<<<2048, 256, 0, stream>>>(gacc, 0.f, (long)Nn * GDc);
  gat_edge_logits<<<eg, 256, 0, stream>>>(ei, ea, meanbuf, xl, xr, g0_we, g0_att,
                                          logits, mx);
  gat_edge_exp<<<eg, 256, 0, stream>>>(ei, logits, mx, exb, den);
  gat_scatter<<<eg, 256, 0, stream>>>(ei, xl, exb, den, gacc);
  gat_finish<<<(int)(((long)Nn * GDc + 255) / 256), 256, 0, stream>>>(gacc, g0_bias, g1b);
  // layer 1: K = 512 (KT=16)
  gemm_bias_act<16><<<dim3(157, 2), 256, 0, stream>>>(g1b, GDc, Nn, g1_wlP, 32,
                                                      g1_bl, xl, (bf16*)nullptr, GDc, 0);
  gemm_bias_act<16><<<dim3(157, 2), 256, 0, stream>>>(g1b, GDc, Nn, g1_wrP, 32,
                                                      g1_br, xr, (bf16*)nullptr, GDc, 0);
  fill_f32<<<256, 256, 0, stream>>>(mx, -__builtin_inff(), (long)Nn * GHc);
  fill_f32<<<256, 256, 0, stream>>>(den, 0.f, (long)Nn * GHc);
  fill_f32<<<2048, 256, 0, stream>>>(gacc, 0.f, (long)Nn * GDc);
  gat_edge_logits<<<eg, 256, 0, stream>>>(ei, ea, meanbuf, xl, xr, g1_we, g1_att,
                                          logits, mx);
  gat_edge_exp<<<eg, 256, 0, stream>>>(ei, logits, mx, exb, den);
  gat_scatter<<<eg, 256, 0, stream>>>(ei, xl, exb, den, gacc);
  gat_finish<<<(int)(((long)Nn * GDc + 255) / 256), 256, 0, stream>>>(gacc, g1_bias, g2b);

  // ---- fusion MLP ----
  concat_fused<<<4096, 256, 0, stream>>>(node, g2b, fusedb);
  gemm_bias_act<24><<<dim3(157, 2), 256, 0, stream>>>(fusedb, HLc + GDc, Nn, fc1P, 32,
                                                      fc1_b, relu1, (bf16*)nullptr, FHc, 1);
  fc2_reduce<<<1250, 256, 0, stream>>>(relu1, fc2_w, fc2_b, out);
}
